// SparseNGCNLayer_57097295233452
// MI455X (gfx1250) — matrix operand order repaired and checked by reading, not compiled, not run on hardware
//
#include <hip/hip_runtime.h>

// CDNA5 / gfx1250, wave32. WMMA fragment types (probe-confirmed builtin signature).
typedef __attribute__((ext_vector_type(16))) _Float16 v16h;
typedef __attribute__((ext_vector_type(8)))  float    v8f;

#define OUT_C 128

__device__ __forceinline__ int imin(int a, int b) { return a < b ? a : b; }

// ---------------------------------------------------------------------------
// Stage 1: sparse-feature SpMM on the matrix unit.
// Each wave takes a group of 16 nonzeros (r_i, c_i, val_i).
//   A (16x32 f16): diagonal, A[i][i] = val_i  (K slots 16..31 = 0)
//   B (32x16 f16): B[k][j]  = W[c_k][16*blk + j]
//   C = A*B  =>  C[i][j] = val_i * W[c_i][16*blk + j]   (f32 accum)
// fp32-accurate via 2-term f16 split: C += Ahi*Bhi + Ahi*Blo + Alo*Bhi.
// Rows of C are scattered with global_atomic_add_f32 (the real bottleneck;
// WMMA cost hides behind it).
//
// Index loads are branchless: full groups use contiguous int4 vector loads
// (clause-able b128s); the tail group clamps indices into range and zeroes
// only the A value, so invalid slots contribute exact +0.0 atomics.
// ---------------------------------------------------------------------------
__global__ __launch_bounds__(256)
void feat_spmm_wmma(const int* __restrict__ fidx,   // [2, nnz] rows then cols
                    const float* __restrict__ fval, // [nnz]
                    const float* __restrict__ W,    // [512, 128] row-major
                    float* __restrict__ out,        // [N, 128], pre-zeroed
                    int nnz)
{
    const int lane  = threadIdx.x & 31;
    const int m     = lane & 15;   // nnz slot / C column index
    const int h     = lane >> 4;   // wave half
    const int wave  = (blockIdx.x * blockDim.x + threadIdx.x) >> 5;
    const int nWav  = (gridDim.x * blockDim.x) >> 5;
    const int nGrp  = (nnz + 15) >> 4;
    const int* __restrict__ frow = fidx;
    const int* __restrict__ fcol = fidx + nnz;

    for (int g = wave; g < nGrp; g += nWav) {
        const int base = g << 4;

        int   c16[16];
        int   r8[8];
        float v;

        if (base + 16 <= nnz) {
            // Full group: contiguous vector loads, no predication.
            const int4* c4 = (const int4*)(fcol + base);
            const int4  c0 = c4[0], c1 = c4[1], c2 = c4[2], c3 = c4[3];
            c16[0]=c0.x;  c16[1]=c0.y;  c16[2]=c0.z;  c16[3]=c0.w;
            c16[4]=c1.x;  c16[5]=c1.y;  c16[6]=c1.z;  c16[7]=c1.w;
            c16[8]=c2.x;  c16[9]=c2.y;  c16[10]=c2.z; c16[11]=c2.w;
            c16[12]=c3.x; c16[13]=c3.y; c16[14]=c3.z; c16[15]=c3.w;
            const int4* r4 = (const int4*)(frow + base + 8 * h);
            const int4  r0 = r4[0], r1 = r4[1];
            r8[0]=r0.x; r8[1]=r0.y; r8[2]=r0.z; r8[3]=r0.w;
            r8[4]=r1.x; r8[5]=r1.y; r8[6]=r1.z; r8[7]=r1.w;
            v = fval[base + m];
        } else {
            // Tail group: clamp indices (always in-bounds), zero invalid values.
            const int last = nnz - 1;
#pragma unroll
            for (int e = 0; e < 16; ++e)  c16[e] = fcol[imin(base + e, last)];
#pragma unroll
            for (int p = 0; p < 8; ++p)   r8[p]  = frow[imin(base + p + 8 * h, last)];
            const int myI = base + m;
            v = fval[imin(myI, last)];
            if (myI > last) v = 0.0f;            // v_cndmask, no branch
        }

        const _Float16 vhi = (_Float16)v;
        const _Float16 vlo = (_Float16)(v - (float)vhi);

        // Diagonal A fragments (hi/lo split).
        // A layout (16-bit A 16x32, contiguous k): low-half lane m holds K0..15 in
        // elements 0..15; high-half lane m holds K16..31 in elements 0..15.
        v16h Ahi = {}; v16h Alo = {};
#pragma unroll
        for (int e = 0; e < 16; ++e) {
            const bool hit = (h == 0) && (e == m);
            Ahi[e] = hit ? vhi : (_Float16)0;
            Alo[e] = hit ? vlo : (_Float16)0;
        }

#pragma unroll
        for (int blk = 0; blk < 8; ++blk) {          // 8 blocks of 16 channels
            // B layout: lanes 0..15 (col j = lane) hold K0..15 in elements
            // 0..15; lanes 16..31 hold K16..31 (zero here).
            v16h Bhi = {}; v16h Blo = {};
            if (h == 0) {
                const int col = (blk << 4) + m;      // j = m
#pragma unroll
                for (int e = 0; e < 16; ++e) {
                    const float w = W[c16[e] * OUT_C + col];
                    const _Float16 whi = (_Float16)w;
                    Bhi[e] = whi;
                    Blo[e] = (_Float16)(w - (float)whi);
                }
            }
            // (neg_a, A, neg_b, B, c_mod, C, reuse_a, reuse_b)
            v8f C = {};
            C = __builtin_amdgcn_wmma_f32_16x16x32_f16(false, Ahi, false, Bhi,
                                                       (short)0, C, false, false);
            C = __builtin_amdgcn_wmma_f32_16x16x32_f16(false, Ahi, false, Blo,
                                                       (short)0, C, false, false);
            C = __builtin_amdgcn_wmma_f32_16x16x32_f16(false, Alo, false, Bhi,
                                                       (short)0, C, false, false);

            // C/D layout: lane l (n=l%16, h=l/16), VGPR p -> row p+8h, col n.
            const int ch = (blk << 4) + m;
#pragma unroll
            for (int p = 0; p < 8; ++p) {
                atomicAdd(&out[r8[p] * OUT_C + ch], C[p]);
            }
        }
    }
}

// ---------------------------------------------------------------------------
// base = relu(base + bias)
// ---------------------------------------------------------------------------
__global__ __launch_bounds__(256)
void bias_relu(float* __restrict__ x, const float* __restrict__ bias, int total)
{
    for (int i = blockIdx.x * blockDim.x + threadIdx.x; i < total;
         i += gridDim.x * blockDim.x) {
        x[i] = fmaxf(x[i] + bias[i & (OUT_C - 1)], 0.0f);
    }
}

// ---------------------------------------------------------------------------
// Stages 2-3: dst[r] += a * src[c]  per edge. One wave per edge; each lane
// handles 4 channels (float4 gather, 4 f32 atomic scatters). Prefetch the
// next edge's source row (global_prefetch_b8) to hide gather latency.
// ---------------------------------------------------------------------------
__global__ __launch_bounds__(256)
void adj_spmm(const int* __restrict__ aidx,   // [2, ne] rows then cols
              const float* __restrict__ aval, // [ne]
              const float* __restrict__ src,  // [N, 128]
              float* __restrict__ dst,        // [N, 128], pre-zeroed
              int ne)
{
    const int tid  = blockIdx.x * blockDim.x + threadIdx.x;
    const int lane = tid & 31;
    const int wave = tid >> 5;
    const int nWav = (gridDim.x * blockDim.x) >> 5;
    const int* __restrict__ ar = aidx;
    const int* __restrict__ ac = aidx + ne;

    for (int e = wave; e < ne; e += nWav) {
        if (e + nWav < ne) {  // prefetch next gathered row for this wave
            __builtin_prefetch(&src[ac[e + nWav] * OUT_C + lane * 4], 0, 1);
        }
        const int   r = ar[e];
        const int   c = ac[e];
        const float a = aval[e];
        const float4 x = *(const float4*)(src + c * OUT_C + lane * 4);
        float* d = dst + r * OUT_C + lane * 4;
        atomicAdd(d + 0, a * x.x);
        atomicAdd(d + 1, a * x.y);
        atomicAdd(d + 2, a * x.z);
        atomicAdd(d + 3, a * x.w);
    }
}

// ---------------------------------------------------------------------------
// Launch: zero(out) -> feat SpMM -> bias+relu -> zero(ws) -> prop(out->ws)
//         -> zero(out) -> prop(ws->out).   ITERATIONS = 3 => 2 propagations.
// Needs one [N,128] f32 scratch buffer (51.2 MB) in d_ws.
// ---------------------------------------------------------------------------
extern "C" void kernel_launch(void* const* d_in, const int* in_sizes, int n_in,
                              void* d_out, int out_size, void* d_ws, size_t ws_size,
                              hipStream_t stream)
{
    (void)n_in; (void)ws_size;
    const int*   fidx = (const int*)d_in[0];
    const float* fval = (const float*)d_in[1];
    const int*   aidx = (const int*)d_in[2];
    const float* aval = (const float*)d_in[3];
    const float* W    = (const float*)d_in[4];
    const float* bias = (const float*)d_in[5];
    float* out = (float*)d_out;
    float* tmp = (float*)d_ws;

    const int nnz = in_sizes[1];   // feat_values count
    const int ne  = in_sizes[3];   // adj_values count
    const size_t bytes = (size_t)out_size * sizeof(float);

    const int blk = 256;
    const int spmmGrid = 2048;     // 16K waves, grid-stride
    const int ewGrid   = (out_size + blk - 1) / blk;

    hipMemsetAsync(out, 0, bytes, stream);
    feat_spmm_wmma<<<spmmGrid, blk, 0, stream>>>(fidx, fval, W, out, nnz);
    bias_relu<<<ewGrid, blk, 0, stream>>>(out, bias, out_size);

    hipMemsetAsync(tmp, 0, bytes, stream);
    adj_spmm<<<spmmGrid, blk, 0, stream>>>(aidx, aval, out, tmp, ne);

    hipMemsetAsync(out, 0, bytes, stream);
    adj_spmm<<<spmmGrid, blk, 0, stream>>>(aidx, aval, tmp, out, ne);
}
